// Encoder_19653770346690
// MI455X (gfx1250) — matrix operand order repaired and checked by reading, not compiled, not run on hardware
//
#include <hip/hip_runtime.h>
#include <hip/hip_bf16.h>
#include <stdint.h>

typedef __attribute__((ext_vector_type(16))) _Float16 v16h;
typedef __attribute__((ext_vector_type(8)))  float    v8f;
typedef _Float16 h16;

#define DEV __device__ __forceinline__

static constexpr int NN = 32;      // nodes
static constexpr int HH = 128;     // hidden
static constexpr int HS = 132;     // padded LDS row stride for h (bank spread)
static constexpr int TT = 50;      // time
static constexpr int BB = 4;       // batch
static constexpr int EE = 992;     // edges (32*31)
static constexpr int BE = BB * EE;         // 3968 sequences
static constexpr int MNE = BB * NN * TT;   // 6400 node-time rows
static constexpr int RR = BE * TT;         // 198400 edge-time rows

DEV float eluf(float x) { return x > 0.f ? x : (__expf(x) - 1.f); }
DEV float sigf(float x) { return 1.f / (1.f + __expf(-x)); }

// A-fragment K offset for 16-bit WMMA 16x32 (element pair base, always even)
DEV int kmapA(int v, int hi) { return 2 * v + (hi ? 16 : 0); }

// ---------------------------------------------------------------------------
// Tensor Data Mover: DMA global -> LDS, tracked with TENSORcnt.
// 5-arg builtin on ROCm 7.2 (clang-22); 6-arg on therock clang-23 (detected
// via its TDM header). Fallback: plain cooperative copy.
// ---------------------------------------------------------------------------
#if defined(__gfx1250__) && __has_builtin(__builtin_amdgcn_tensor_load_to_lds)
#define USE_TDM 1
#if __has_include(<hip/amd_detail/amd_gfx1250_TDM.h>)
#define TDM_ARGS6 1
#endif
#endif

#ifdef USE_TDM
typedef __attribute__((ext_vector_type(4))) unsigned int v4u;
typedef __attribute__((ext_vector_type(4))) int v4i;
typedef __attribute__((ext_vector_type(8))) int v8i;

// 1-row tile, data_size = 8 bytes, `bytes` must be a multiple of 8 (<512KB).
DEV void tdm_load_to_lds(const void* gsrc, uint32_t lds_off, uint32_t bytes) {
  uint64_t ga = (uint64_t)(uintptr_t)gsrc;
  uint32_t u = bytes >> 3;  // 8-byte units (12288 for 96KB: fits 16-bit tile_dim0)
  // D# group 0: count=1 | lds_addr | global_addr[56:0] | type=2
  v4u g0 = { 1u, lds_off, (uint32_t)ga,
             (uint32_t)((ga >> 32) & 0x01FFFFFFu) | 0x80000000u };
  // D# group 1: data_size=3(8B); tensor_dim0=u; tensor_dim1=1; tile_dim0=u;
  //             tile_dim1=1; tensor_dim0_stride=u
  v8i g1 = { (int)0x00030000,
             (int)(u << 16),      // tensor_dim0[15:0] at bits 63:48
             (int)0x00010000,     // tensor_dim1 = 1 (bits 111:80)
             (int)(u << 16),      // tile_dim0 at bits 127:112
             1,                   // tile_dim1 = 1
             (int)u,              // tensor_dim0_stride[31:0]
             0, 0 };
  v4i z4 = {0, 0, 0, 0};
#ifdef TDM_ARGS6
  v8i z8 = {0, 0, 0, 0, 0, 0, 0, 0};
  __builtin_amdgcn_tensor_load_to_lds(g0, g1, z4, z4, z8, 0);
#else
  __builtin_amdgcn_tensor_load_to_lds(g0, g1, z4, z4, 0);
#endif
}
#endif

// ---------------------------------------------------------------------------
// K0: pack weights to f16; GRU + W_es2 weights into WMMA B-fragment order.
// B-frag order: tile = nt*KT + kt ; within tile: [lane][16 halves], where
// lane<16 -> N = nt*16+lane, K = kt*32 + e ; lane>=16 -> K = kt*32+16+e.
// ---------------------------------------------------------------------------
__global__ void pack_kernel(const float* __restrict__ Wne2, const float* __restrict__ Wec,
                            const float* __restrict__ Wes2,
                            const float* __restrict__ WihF, const float* __restrict__ WhhF,
                            const float* __restrict__ WihR, const float* __restrict__ WhhR,
                            h16* __restrict__ w2h, h16* __restrict__ wecA, h16* __restrict__ wecB,
                            h16* __restrict__ wes2p,
                            h16* __restrict__ pwihF, h16* __restrict__ pwhhF,
                            h16* __restrict__ pwihR, h16* __restrict__ pwhhR) {
  int idx = blockIdx.x * blockDim.x + threadIdx.x;
  if (idx < 16384) { w2h[idx] = (h16)Wne2[idx]; return; }
  idx -= 16384;
  if (idx < 16384) {   // W_ec split: Wa - Wb and Wb
    int hh = idx >> 7, k = idx & 127;
    float a = Wec[hh * 256 + k], b = Wec[hh * 256 + 128 + k];
    wecA[idx] = (h16)(a - b);
    wecB[idx] = (h16)b;
    return;
  }
  idx -= 16384;
  if (idx < 32768) {   // W_es2 packed: N=128 (8 ntiles), K=256 (8 ktiles)
    int tile = idx >> 9, r = (idx >> 4) & 31, e = idx & 15;
    int nt = tile >> 3, kt = tile & 7;
    int n = nt * 16 + (r & 15), k = kt * 32 + ((r >= 16) ? 16 : 0) + e;
    wes2p[idx] = (h16)Wes2[n * 256 + k];
    return;
  }
  idx -= 32768;
  if (idx < 4 * 49152) {  // GRU weights packed: N=384 (24 ntiles), K=128 (4 ktiles)
    int w = idx / 49152, o = idx % 49152;
    int tile = o >> 9, r = (o >> 4) & 31, e = o & 15;
    int nt = tile >> 2, kt = tile & 3;
    int n = nt * 16 + (r & 15), k = kt * 32 + ((r >= 16) ? 16 : 0) + e;
    const float* src = (w == 0) ? WihF : (w == 1) ? WhhF : (w == 2) ? WihR : WhhR;
    h16* dst = (w == 0) ? pwihF : (w == 1) ? pwhhF : (w == 2) ? pwihR : pwhhR;
    dst[o] = (h16)src[n * 128 + k];
  }
}

// ---------------------------------------------------------------------------
// K1: h1 = f16(elu(x @ W_ne1.T + b_ne1)); x[b,n,t,:] = inputs[b,t,n,:], INP=4
// ---------------------------------------------------------------------------
__global__ void h1_kernel(const float* __restrict__ inp, const float* __restrict__ W1,
                          const float* __restrict__ b1, h16* __restrict__ h1) {
  int idx = blockIdx.x * blockDim.x + threadIdx.x;
  if (idx >= MNE * HH) return;
  int h = idx & 127;
  int row = idx >> 7;
  int b = row / (NN * TT);
  int rem = row % (NN * TT);
  int n = rem / TT, t = rem % TT;
  const float* xp = inp + (((size_t)b * TT + t) * NN + n) * 4;
  const float* wp = W1 + h * 4;
  float a = b1[h] + xp[0] * wp[0] + xp[1] * wp[1] + xp[2] * wp[2] + xp[3] * wp[3];
  h1[idx] = (h16)eluf(a);
}

// ---------------------------------------------------------------------------
// K2: generic WMMA GEMM  out = A(f16, MxK rowmajor) @ W(f16, NxK rowmajor).T
// ---------------------------------------------------------------------------
__global__ void gemm_wmma(const h16* __restrict__ A, const h16* __restrict__ W,
                          const float* __restrict__ bias,
                          float* __restrict__ outF, h16* __restrict__ outH,
                          int M, int Nn, int K) {
  int wave = (blockIdx.x * blockDim.x + threadIdx.x) >> 5;
  int tilesN = Nn >> 4;
  int nTiles = (M >> 4) * tilesN;
  if (wave >= nTiles) return;
  int m0 = (wave / tilesN) * 16, n0 = (wave % tilesN) * 16;
  int lane = threadIdx.x & 31, lr = lane & 15, hi = (lane >= 16) ? 1 : 0;
  const v8f vz = {0.f, 0.f, 0.f, 0.f, 0.f, 0.f, 0.f, 0.f};
  v8f c = vz;
  const uint32_t* Arow = (const uint32_t*)(A + (size_t)(m0 + lr) * K);
  const uint32_t* Wrow = (const uint32_t*)(W + (size_t)(n0 + lr) * K);
  for (int k0 = 0; k0 < K; k0 += 32) {
    v16h a, b;
    uint32_t* ap = (uint32_t*)&a;
    uint32_t* bp = (uint32_t*)&b;
#pragma unroll
    for (int v = 0; v < 8; ++v) {
      int ka = k0 + kmapA(v, hi);
      ap[v] = Arow[ka >> 1];
      int kb = k0 + (hi ? 16 : 0) + 2 * v;
      bp[v] = Wrow[kb >> 1];
    }
    c = __builtin_amdgcn_wmma_f32_16x16x32_f16(false, a, false, b, (short)0, c, false, false);
  }
  float bv = bias ? bias[n0 + lr] : 0.f;
  float* cp = (float*)&c;
#pragma unroll
  for (int v = 0; v < 8; ++v) {
    int row = m0 + v + hi * 8;
    float val = cp[v] + bv;
    if (outF) outF[(size_t)row * Nn + n0 + lr] = val;
    if (outH) outH[(size_t)row * Nn + n0 + lr] = (h16)val;
  }
}

// ---------------------------------------------------------------------------
// K3: node_embeddings[b,d,t,h] = P[b,d,t,h] + b_ec[h] + max_{s != d} Q[b,s,t,h]
// ---------------------------------------------------------------------------
__global__ void agg_kernel(const float* __restrict__ P, const float* __restrict__ Q,
                           const float* __restrict__ bec, float* __restrict__ nemb) {
  int idx = blockIdx.x * blockDim.x + threadIdx.x;
  if (idx >= BB * TT * HH) return;
  int h = idx & 127;
  int bt = idx >> 7;
  int t = bt % TT, b = bt / TT;
  float m1 = -3.4e38f, m2 = -3.4e38f;
  int arg = -1;
  for (int n = 0; n < NN; ++n) {
    float q = Q[(((size_t)b * NN + n) * TT + t) * HH + h];
    if (q > m1) { m2 = m1; m1 = q; arg = n; }
    else if (q > m2) { m2 = q; }
  }
  float be = bec[h];
  for (int d = 0; d < NN; ++d) {
    size_t o = (((size_t)b * NN + d) * TT + t) * HH + h;
    float mx = (d == arg) ? m2 : m1;
    nemb[o] = P[o] + be + mx;
  }
}

// ---------------------------------------------------------------------------
// K4: dots[(b*EE+e)*TT + t] = <nemb[b,src,t,:], nemb[b,dst,t,:]>/H
//     LDS row stride 129 -> conflict-free row-parallel reads
// ---------------------------------------------------------------------------
__global__ void dots_kernel(const float* __restrict__ nemb, const int* __restrict__ ei,
                            float* __restrict__ dots) {
  __shared__ float s[NN * 129];
  int b = blockIdx.x / TT, t = blockIdx.x % TT;
  for (int i = threadIdx.x; i < NN * HH; i += blockDim.x) {
    int n = i >> 7, h = i & 127;
    s[n * 129 + h] = nemb[(((size_t)b * NN + n) * TT + t) * HH + h];
  }
  __syncthreads();
  for (int p = threadIdx.x; p < EE; p += blockDim.x) {
    int src = ei[p], dst = ei[EE + p];
    const float* a = s + src * 129;
    const float* c = s + dst * 129;
    float d0 = 0.f, d1 = 0.f, d2 = 0.f, d3 = 0.f;
#pragma unroll
    for (int k = 0; k < HH; k += 4) {
      d0 += a[k] * c[k];
      d1 += a[k + 1] * c[k + 1];
      d2 += a[k + 2] * c[k + 2];
      d3 += a[k + 3] * c[k + 3];
    }
    dots[((size_t)b * EE + p) * TT + t] = ((d0 + d1) + (d2 + d3)) * (1.f / 128.f);
  }
}

// ---------------------------------------------------------------------------
// K5: ef = elu(d * W_es1 + b_es1) @ W_es2.T + b_es2 ; fused rank-1 A build.
// ---------------------------------------------------------------------------
__global__ void ef_kernel(const float* __restrict__ dots, const float* __restrict__ Wes1,
                          const float* __restrict__ bes1, const h16* __restrict__ wes2p,
                          const float* __restrict__ bes2, h16* __restrict__ ef) {
  __shared__ float sw1[256], sb1[256], sb2[128];
  for (int i = threadIdx.x; i < 256; i += blockDim.x) { sw1[i] = Wes1[i]; sb1[i] = bes1[i]; }
  for (int i = threadIdx.x; i < 128; i += blockDim.x) sb2[i] = bes2[i];
  __syncthreads();
  int wave = (blockIdx.x * blockDim.x + threadIdx.x) >> 5;
  if (wave >= RR / 16) return;
  int m0 = wave * 16;
  int lane = threadIdx.x & 31, lr = lane & 15, hi = (lane >= 16) ? 1 : 0;
  float d = dots[m0 + lr];
  const v8f vz = {0.f, 0.f, 0.f, 0.f, 0.f, 0.f, 0.f, 0.f};
  v8f acc[8];
#pragma unroll
  for (int i = 0; i < 8; ++i) acc[i] = vz;
  for (int kt = 0; kt < 8; ++kt) {
    int k0 = kt * 32;
    v16h a;
    _Float16* ap = (_Float16*)&a;
#pragma unroll
    for (int v = 0; v < 8; ++v) {
      int ka = k0 + kmapA(v, hi);
      ap[2 * v]     = (h16)eluf(d * sw1[ka] + sb1[ka]);
      ap[2 * v + 1] = (h16)eluf(d * sw1[ka + 1] + sb1[ka + 1]);
    }
#pragma unroll
    for (int nt = 0; nt < 8; ++nt) {
      v16h b = *(const v16h*)(wes2p + ((size_t)(nt * 8 + kt) * 32 + lane) * 16);
      acc[nt] = __builtin_amdgcn_wmma_f32_16x16x32_f16(false, a, false, b, (short)0, acc[nt], false, false);
    }
  }
#pragma unroll
  for (int nt = 0; nt < 8; ++nt) {
    float bv = sb2[nt * 16 + lr];
    float* cp = (float*)&acc[nt];
#pragma unroll
    for (int v = 0; v < 8; ++v) {
      int row = m0 + v + hi * 8;
      ef[(size_t)row * HH + nt * 16 + lr] = (h16)(cp[v] + bv);
    }
  }
}

// ---------------------------------------------------------------------------
// K6: persistent GRU. 62 blocks x 4 waves; wave owns 16 sequences.
// TDM stages both weight matrices (192 KB) into LDS; fp32 h double-buffered
// in LDS with padded stride; 24 WMMAs/step; NE=2 head folded in.
// ---------------------------------------------------------------------------
__global__ void gru_kernel(const h16* __restrict__ ef,
                           const h16* __restrict__ pwih, const h16* __restrict__ pwhh,
                           const float* __restrict__ bih, const float* __restrict__ bhh,
                           const float* __restrict__ Whead, const float* __restrict__ bhead,
                           const h16* __restrict__ fwdx_in,  // reverse pass input
                           h16* __restrict__ fwdx_out,       // forward pass output
                           float* __restrict__ head_out,     // prior or encoder (B,T,E,NE)
                           float* __restrict__ pstate,       // forward only
                           int reverse) {
  extern __shared__ char smem[];
  h16* sWih = (h16*)smem;               // 384*128 halves = 96 KB  (LDS offset 0)
  h16* sWhh = sWih + 49152;             // 96 KB                   (LDS offset 98304)
  float* hbuf = (float*)(sWhh + 49152); // 4 waves * 2 * 16*HS fp32
#ifdef USE_TDM
  if (threadIdx.x == 0) {
    tdm_load_to_lds(pwih, 0u, 98304u);
    tdm_load_to_lds(pwhh, 98304u, 98304u);
  }
#else
  for (int i = threadIdx.x; i < 49152; i += blockDim.x) {
    sWih[i] = pwih[i];
    sWhh[i] = pwhh[i];
  }
#endif
  int wv = threadIdx.x >> 5;
  int lane = threadIdx.x & 31, lr = lane & 15, hi = (lane >= 16) ? 1 : 0;
  int m0 = blockIdx.x * 64 + wv * 16;
  float* hA = hbuf + wv * (2 * 16 * HS);
  float* hB = hA + 16 * HS;
  for (int i = lane; i < 16 * HS; i += 32) hA[i] = 0.f;
#ifdef USE_TDM
  __builtin_amdgcn_s_wait_tensorcnt(0);
#endif
  __syncthreads();
  float* hcur = hA;
  float* hnxt = hB;
  const v8f vz = {0.f, 0.f, 0.f, 0.f, 0.f, 0.f, 0.f, 0.f};

  for (int s = 0; s < TT; ++s) {
    int t = reverse ? (TT - 1 - s) : s;
    // A fragments: x from ef (global f16), h from LDS fp32 carry
    v16h ax[4], ah[4];
    const h16* efrow = ef + ((size_t)(m0 + lr) * TT + t) * HH;
#pragma unroll
    for (int kt = 0; kt < 4; ++kt) {
      uint32_t* axp = (uint32_t*)&ax[kt];
      _Float16* ahp = (_Float16*)&ah[kt];
#pragma unroll
      for (int v = 0; v < 8; ++v) {
        int ka = kt * 32 + kmapA(v, hi);
        axp[v] = *(const uint32_t*)(efrow + ka);
        ahp[2 * v]     = (h16)hcur[lr * HS + ka];
        ahp[2 * v + 1] = (h16)hcur[lr * HS + ka + 1];
      }
    }
#pragma unroll 1
    for (int j = 0; j < 8; ++j) {
      v8f accx[3], acch[3];
#pragma unroll
      for (int g = 0; g < 3; ++g) {
        accx[g] = vz;
        acch[g] = vz;
        int nt = g * 8 + j;
#pragma unroll
        for (int kt = 0; kt < 4; ++kt) {
          v16h bx = *(const v16h*)(sWih + ((size_t)(nt * 4 + kt) * 32 + lane) * 16);
          accx[g] = __builtin_amdgcn_wmma_f32_16x16x32_f16(false, ax[kt], false, bx, (short)0, accx[g], false, false);
          v16h bh = *(const v16h*)(sWhh + ((size_t)(nt * 4 + kt) * 32 + lane) * 16);
          acch[g] = __builtin_amdgcn_wmma_f32_16x16x32_f16(false, ah[kt], false, bh, (short)0, acch[g], false, false);
        }
      }
      int c = j * 16 + lr;
      float bxr = bih[c], bxz = bih[c + 128], bxn = bih[c + 256];
      float bhr = bhh[c], bhz = bhh[c + 128], bhn = bhh[c + 256];
      float* xr = (float*)&accx[0]; float* xz = (float*)&accx[1]; float* xn = (float*)&accx[2];
      float* hr = (float*)&acch[0]; float* hz = (float*)&acch[1]; float* hn = (float*)&acch[2];
#pragma unroll
      for (int v = 0; v < 8; ++v) {
        int rloc = v + hi * 8;
        float r = sigf(xr[v] + bxr + hr[v] + bhr);
        float z = sigf(xz[v] + bxz + hz[v] + bhz);
        float n = tanhf(xn[v] + bxn + r * (hn[v] + bhn));
        float hold = hcur[rloc * HS + c];
        hnxt[rloc * HS + c] = (1.f - z) * n + z * hold;
      }
    }
    __syncthreads();
    // NE=2 head: lane -> (row = lane&15, ch = lane>>4)
    {
      int r16 = lr, ch = hi;
      size_t seqrow = (size_t)m0 + r16;
      int b = (int)(seqrow / EE), e = (int)(seqrow % EE);
      size_t off = (((size_t)b * TT + t) * EE + e) * 2 + ch;
      const float* hrow = hnxt + r16 * HS;
      if (!reverse) {
        const float* wrow = Whead + ch * HH;
        float d0 = 0.f, d1 = 0.f, d2 = 0.f, d3 = 0.f;
#pragma unroll
        for (int k = 0; k < HH; k += 4) {
          d0 += hrow[k] * wrow[k];
          d1 += hrow[k + 1] * wrow[k + 1];
          d2 += hrow[k + 2] * wrow[k + 2];
          d3 += hrow[k + 3] * wrow[k + 3];
        }
        head_out[off] = bhead[ch] + ((d0 + d1) + (d2 + d3));
        // packed f16 stores of fwd_x (2 halves / b32)
        for (int i = lane; i < 1024; i += 32) {
          int rl = i >> 6, kp = (i & 63) << 1;
          h16 h0 = (h16)hnxt[rl * HS + kp];
          h16 h1 = (h16)hnxt[rl * HS + kp + 1];
          uint16_t u0, u1;
          __builtin_memcpy(&u0, &h0, 2);
          __builtin_memcpy(&u1, &h1, 2);
          *(uint32_t*)(fwdx_out + ((size_t)(m0 + rl) * TT + t) * HH + kp) =
              (uint32_t)u0 | ((uint32_t)u1 << 16);
        }
      } else {
        const float* wrow = Whead + ch * 2 * HH;
        const h16* fx = fwdx_in + (seqrow * TT + t) * HH;
        float d0 = 0.f, d1 = 0.f;
#pragma unroll
        for (int k = 0; k < HH; k += 2) {
          d0 += (float)fx[k] * wrow[k] + hrow[k] * wrow[HH + k];
          d1 += (float)fx[k + 1] * wrow[k + 1] + hrow[k + 1] * wrow[HH + k + 1];
        }
        head_out[off] = bhead[ch] + d0 + d1;
      }
    }
    float* tmpp = hcur; hcur = hnxt; hnxt = tmpp;
  }
  if (!reverse) {
    for (int i = lane; i < 512; i += 32) {
      int rl = i >> 5, kq = (i & 31) << 2;
      const float* hp = hcur + rl * HS + kq;
      float4 vv = make_float4(hp[0], hp[1], hp[2], hp[3]);
      *(float4*)(pstate + (size_t)(m0 + rl) * HH + kq) = vv;
    }
  }
}

// ---------------------------------------------------------------------------
extern "C" void kernel_launch(void* const* d_in, const int* in_sizes, int n_in,
                              void* d_out, int out_size, void* d_ws, size_t ws_size,
                              hipStream_t stream) {
  (void)in_sizes; (void)n_in; (void)out_size; (void)ws_size;
  const float* inputs = (const float*)d_in[0];
  const int*   ei     = (const int*)d_in[1];
  const float* W_ne1  = (const float*)d_in[2];
  const float* b_ne1  = (const float*)d_in[3];
  const float* W_ne2  = (const float*)d_in[4];
  const float* b_ne2  = (const float*)d_in[5];
  const float* W_ec   = (const float*)d_in[6];
  const float* b_ec   = (const float*)d_in[7];
  const float* W_es1  = (const float*)d_in[8];
  const float* b_es1  = (const float*)d_in[9];
  const float* W_es2  = (const float*)d_in[10];
  const float* b_es2  = (const float*)d_in[11];
  const float* Wih_f  = (const float*)d_in[12];
  const float* Whh_f  = (const float*)d_in[13];
  const float* bih_f  = (const float*)d_in[14];
  const float* bhh_f  = (const float*)d_in[15];
  const float* Wih_r  = (const float*)d_in[16];
  const float* Whh_r  = (const float*)d_in[17];
  const float* bih_r  = (const float*)d_in[18];
  const float* bhh_r  = (const float*)d_in[19];
  const float* W_pf   = (const float*)d_in[20];
  const float* b_pf   = (const float*)d_in[21];
  const float* W_ef   = (const float*)d_in[22];
  const float* b_ef   = (const float*)d_in[23];

  float* out    = (float*)d_out;
  float* prior  = out;                      // (B,T,E,NE)  396800
  float* enc    = out + 396800;             // (B,T,E,NE)  396800
  float* pstate = out + 793600;             // (1,B*E,HR)  507904
  float* nemb   = out + 1301504;            // (B,N,T,H)   819200

  char* ws = (char*)d_ws;
  size_t o = 0;
  h16* w2h   = (h16*)(ws + o); o += 32768;      // 128x128
  h16* wecA  = (h16*)(ws + o); o += 32768;
  h16* wecB  = (h16*)(ws + o); o += 32768;
  h16* wes2p = (h16*)(ws + o); o += 65536;      // 128x256 packed
  h16* pwihF = (h16*)(ws + o); o += 98304;      // 384x128 packed
  h16* pwhhF = (h16*)(ws + o); o += 98304;
  h16* pwihR = (h16*)(ws + o); o += 98304;
  h16* pwhhR = (h16*)(ws + o); o += 98304;
  h16* h1h   = (h16*)(ws + o); o += (size_t)MNE * HH * 2;   // 1.6 MB
  h16* neh   = (h16*)(ws + o); o += (size_t)MNE * HH * 2;   // 1.6 MB
  float* Pf  = (float*)(ws + o); o += (size_t)MNE * HH * 4; // 3.3 MB
  float* Qf  = (float*)(ws + o); o += (size_t)MNE * HH * 4; // 3.3 MB
  float* dts = (float*)(ws + o); o += (size_t)RR * 4;       // 0.8 MB
  h16* efh   = (h16*)(ws + o); o += (size_t)RR * HH * 2;    // 50.8 MB
  h16* fwdxh = (h16*)(ws + o); o += (size_t)RR * HH * 2;    // 50.8 MB

  constexpr uint32_t GRU_LDS = 2u * 49152u * 2u + 4u * 2u * 16u * HS * 4u; // 264192 B

  // K0: weight packing (262144 elements)
  pack_kernel<<<1024, 256, 0, stream>>>(W_ne2, W_ec, W_es2, Wih_f, Whh_f, Wih_r, Whh_r,
                                        w2h, wecA, wecB, wes2p, pwihF, pwhhF, pwihR, pwhhR);
  // K1: node MLP layer 1 (K=4, scalar)
  h1_kernel<<<(MNE * HH + 255) / 256, 256, 0, stream>>>(inputs, W_ne1, b_ne1, h1h);
  // K2: ne = h1 @ W_ne2.T + b ; P = ne @ (Wa-Wb).T ; Q = ne @ Wb.T
  gemm_wmma<<<400, 256, 0, stream>>>(h1h, w2h, b_ne2, nullptr, neh, MNE, HH, HH);
  gemm_wmma<<<400, 256, 0, stream>>>(neh, wecA, nullptr, Pf, nullptr, MNE, HH, HH);
  gemm_wmma<<<400, 256, 0, stream>>>(neh, wecB, nullptr, Qf, nullptr, MNE, HH, HH);
  // K3: segment_max via top-2 -> node_embeddings (direct to d_out)
  agg_kernel<<<(BB * TT * HH + 255) / 256, 256, 0, stream>>>(Pf, Qf, b_ec, nemb);
  // K4: pairwise dots
  dots_kernel<<<BB * TT, 256, 0, stream>>>(nemb, ei, dts);
  // K5: fused edge-score MLP -> ef (f16)
  ef_kernel<<<1550, 256, 0, stream>>>(dts, W_es1, b_es1, wes2p, b_es2, efh);
  // K6: persistent bidirectional GRU + heads (TDM weight staging, 258 KB LDS)
  gru_kernel<<<62, 128, GRU_LDS, stream>>>(efh, pwihF, pwhhF, bih_f, bhh_f,
                                           W_pf, b_pf, nullptr, fwdxh, prior, pstate, 0);
  gru_kernel<<<62, 128, GRU_LDS, stream>>>(efh, pwihR, pwhhR, bih_r, bhh_r,
                                           W_ef, b_ef, fwdxh, nullptr, enc, nullptr, 1);
}
